// DeformConvTranspose_55207509623210
// MI455X (gfx1250) — matrix operand order repaired and checked by reading, not compiled, not run on hardware
//
#include <hip/hip_runtime.h>

// Problem constants (match reference)
#define B_N   8
#define CIN   128
#define H_IN  64
#define W_IN  64
#define COUT  128
#define KH    4
#define KW    4
#define KK    16                    // KH*KW
#define HO    61
#define WO    61
#define NPIX  (HO*WO)               // 3721
#define CK    (CIN*KK)              // 2048 reduction dimension (ck = c*16 + k)
#define NSLAB (CK/32)               // 64 K-slabs of 32
#define CHUNK 256                   // ck per LDS chunk (16 input channels)
#define NCHUNK (CK/CHUNK)           // 8
#define SLABS_PER_CHUNK (CHUNK/32)  // 8
#define PT    32                    // pixels per tile (N dim, two 16-wide WMMA tiles)
#define NTILE ((NPIX + PT - 1)/PT)  // 117
#define PROW  132                   // packed LDS row stride in dwords (CHUNK/2 + 4 pad)

typedef __attribute__((ext_vector_type(16))) __bf16       v16bf;
typedef __attribute__((ext_vector_type(8)))  float        v8f;
typedef __attribute__((ext_vector_type(8)))  unsigned int v8u;

__device__ __forceinline__ unsigned bf16_rne(float f) {
  unsigned u = __float_as_uint(f);
  return (u + 0x7FFFu + ((u >> 16) & 1u)) >> 16;
}

// f32 -> (bf16 hi, bf16 lo) with hi+lo ~ f to ~2^-18 relative
__device__ __forceinline__ void bf16_split(float f, unsigned& hi, unsigned& lo) {
  hi = bf16_rne(f);
  float fh = __uint_as_float(hi << 16);
  lo = bf16_rne(f - fh);
}

// ---------------------------------------------------------------------------
// Kernel 0: flip weights, split bf16 hi/lo, store in exact A-fragment layout:
//   whi[((m*NSLAB + s)*32 + lane)*16 + elem]
//   lane 0-15 : M = lane,    elems 0-15 -> K 0..15
//   lane16-31 : M = lane-16, elems 0-15 -> K 16..31
// ---------------------------------------------------------------------------
__global__ void prep_weights(const float* __restrict__ weight,
                             unsigned short* __restrict__ whi,
                             unsigned short* __restrict__ wlo) {
  int tid  = blockIdx.x * 256 + threadIdx.x;   // 0 .. 262143
  int elem = tid & 15;
  int lane = (tid >> 4) & 31;
  int s    = (tid >> 9) & 63;
  int m    = tid >> 15;
  int rowM = lane & 15;
  int half = lane >> 4;
  int k32  = half * 16 + elem;
  int ck = s * 32 + k32;
  int c  = ck >> 4;
  int k  = ck & 15;
  int ki = k >> 2, kj = k & 3;
  int cout = m * 16 + rowM;
  float w = weight[((cout * CIN + c) * KH + (KH - 1 - ki)) * KW + (KW - 1 - kj)];
  unsigned hi, lo;
  bf16_split(w, hi, lo);
  whi[tid] = (unsigned short)hi;
  wlo[tid] = (unsigned short)lo;
}

// ---------------------------------------------------------------------------
// Main fused kernel: bilinear gather -> pre-paired bf16 hi/lo in LDS -> WMMA.
// One workgroup (8 waves, 256 thr) per (batch, 32-pixel tile).
// Wave m computes couts [16m,16m+16) x 32 pixels (2 accumulators).
// Sampling: thread owns k-pair (2kp, 2kp+1) at one pixel; writes packed
// bf16 pairs straight into Shi/Slo -> WMMA B fragments need zero repacking.
// ---------------------------------------------------------------------------
__global__ __launch_bounds__(256)
void deform_conv_wmma(const float* __restrict__ inp,
                      const float* __restrict__ off,
                      const float* __restrict__ msk,
                      const float* __restrict__ bias,
                      const unsigned short* __restrict__ whi,
                      const unsigned short* __restrict__ wlo,
                      float* __restrict__ out) {
  __shared__ unsigned int Shi[PT * PROW];   // packed {hi(2j+1),hi(2j)} per dword
  __shared__ unsigned int Slo[PT * PROW];

  const int tile = blockIdx.x % NTILE;
  const int b    = blockIdx.x / NTILE;
  const int p0   = tile * PT;

  const int t  = threadIdx.x;
  const int kp = t >> 5;      // k-pair 0..7  (k = 2kp, 2kp+1)
  const int pl = t & 31;      // pixel-in-tile 0..31
  const int p  = p0 + pl;

  // ---- per-thread geometry for both taps (registers, reused all chunks) ----
  int   gidx[8];
  float gwt[8];
  #pragma unroll
  for (int kk = 0; kk < 2; ++kk) {
    const int k = kp * 2 + kk;
    int i0 = 0, i1 = 0, i2 = 0, i3 = 0;
    float a0 = 0.f, a1 = 0.f, a2 = 0.f, a3 = 0.f;
    if (p < NPIX) {
      const int ho = p / WO;
      const int wo = p - ho * WO;
      const int ki = k >> 2, kj = k & 3;
      const size_t ob = ((size_t)b * (2 * KK) + 2 * k) * NPIX + p;
      const float dy = off[ob];
      const float dx = off[ob + NPIX];
      const float mk = msk[((size_t)b * KK + k) * NPIX + p];
      const float y  = dy + (float)(ki + ho);
      const float x  = dx + (float)(kj + wo);
      const float y0f = floorf(y), x0f = floorf(x);
      const int   y0  = (int)y0f,  x0  = (int)x0f;
      const float wy = y - y0f, wx = x - x0f;
      const float vy0 = (y0     >= 0 && y0     < H_IN) ? 1.f : 0.f;
      const float vy1 = (y0 + 1 >= 0 && y0 + 1 < H_IN) ? 1.f : 0.f;
      const float vx0 = (x0     >= 0 && x0     < W_IN) ? 1.f : 0.f;
      const float vx1 = (x0 + 1 >= 0 && x0 + 1 < W_IN) ? 1.f : 0.f;
      const int cy0 = min(max(y0,     0), H_IN - 1);
      const int cy1 = min(max(y0 + 1, 0), H_IN - 1);
      const int cx0 = min(max(x0,     0), W_IN - 1);
      const int cx1 = min(max(x0 + 1, 0), W_IN - 1);
      i0 = cy0 * W_IN + cx0;  i1 = cy0 * W_IN + cx1;
      i2 = cy1 * W_IN + cx0;  i3 = cy1 * W_IN + cx1;
      a0 = (1.f - wy) * (1.f - wx) * vy0 * vx0 * mk;
      a1 = (1.f - wy) * wx         * vy0 * vx1 * mk;
      a2 = wy         * (1.f - wx) * vy1 * vx0 * mk;
      a3 = wy         * wx         * vy1 * vx1 * mk;
    }
    gidx[kk * 4 + 0] = i0; gidx[kk * 4 + 1] = i1;
    gidx[kk * 4 + 2] = i2; gidx[kk * 4 + 3] = i3;
    gwt[kk * 4 + 0] = a0;  gwt[kk * 4 + 1] = a1;
    gwt[kk * 4 + 2] = a2;  gwt[kk * 4 + 3] = a3;
  }

  const float* inpb = inp + (size_t)b * CIN * H_IN * W_IN;

  const int lane  = t & 31;
  const int m     = t >> 5;       // wave id == M tile (cout/16)
  const int bhalf = lane >> 4;    // K-half within slab for B/D layout
  const int bp    = lane & 15;    // B-fragment column

  v8f acc0 = {0.f, 0.f, 0.f, 0.f, 0.f, 0.f, 0.f, 0.f};  // pixels p0 .. p0+15
  v8f acc1 = {0.f, 0.f, 0.f, 0.f, 0.f, 0.f, 0.f, 0.f};  // pixels p0+16 .. p0+31

  for (int cc = 0; cc < NCHUNK; ++cc) {
    // ---- sampling: Shi/Slo[pl][i*8 + kp] = packed bf16 pair for c=cc*16+i ----
    const int c0 = cc * 16;
    #pragma unroll
    for (int i = 0; i < 16; ++i) {
      const float* pc = inpb + (size_t)(c0 + i) * (H_IN * W_IN);
      float ve = gwt[0] * pc[gidx[0]] + gwt[1] * pc[gidx[1]] +
                 gwt[2] * pc[gidx[2]] + gwt[3] * pc[gidx[3]];
      float vo = gwt[4] * pc[gidx[4]] + gwt[5] * pc[gidx[5]] +
                 gwt[6] * pc[gidx[6]] + gwt[7] * pc[gidx[7]];
      unsigned he, le, ho_, lo_;
      bf16_split(ve, he, le);
      bf16_split(vo, ho_, lo_);
      const int addr = pl * PROW + i * 8 + kp;
      Shi[addr] = he | (ho_ << 16);   // even tap in [15:0], odd in [31:16]
      Slo[addr] = le | (lo_ << 16);
    }
    __syncthreads();

    // ---- WMMA: 8 slabs of K=32, two 16-pixel N tiles ----
    #pragma unroll
    for (int s = 0; s < SLABS_PER_CHUNK; ++s) {
      const int sg = cc * SLABS_PER_CHUNK + s;

      // A fragments (precomputed layout): 32B hi + 32B lo per lane
      const size_t abase = ((size_t)(m * NSLAB + sg) * 32 + lane) * 16;
      const uint4* ph = (const uint4*)(whi + abase);
      const uint4* pL = (const uint4*)(wlo + abase);
      uint4 h0 = ph[0], h1 = ph[1];
      uint4 l0 = pL[0], l1 = pL[1];
      v8u ahraw = {h0.x, h0.y, h0.z, h0.w, h1.x, h1.y, h1.z, h1.w};
      v8u alraw = {l0.x, l0.y, l0.z, l0.w, l1.x, l1.y, l1.z, l1.w};
      v16bf Ahi = __builtin_bit_cast(v16bf, ahraw);
      v16bf Alo = __builtin_bit_cast(v16bf, alraw);

      // B fragments read directly (pre-paired): 8 dwords = 16 bf16 per lane
      const int bbase = s * 16 + bhalf * 8;          // dword index within row

      {
        const uint4* rh = (const uint4*)&Shi[bp * PROW + bbase];
        const uint4* rl = (const uint4*)&Slo[bp * PROW + bbase];
        uint4 x0 = rh[0], x1 = rh[1];
        uint4 y0 = rl[0], y1 = rl[1];
        v8u bh = {x0.x, x0.y, x0.z, x0.w, x1.x, x1.y, x1.z, x1.w};
        v8u bl = {y0.x, y0.y, y0.z, y0.w, y1.x, y1.y, y1.z, y1.w};
        v16bf Bhi = __builtin_bit_cast(v16bf, bh);
        v16bf Blo = __builtin_bit_cast(v16bf, bl);
        acc0 = __builtin_amdgcn_wmma_f32_16x16x32_bf16(false, Ahi, false, Bhi,
                                                       (short)0, acc0, false, false);
        acc0 = __builtin_amdgcn_wmma_f32_16x16x32_bf16(false, Ahi, false, Blo,
                                                       (short)0, acc0, false, false);
        acc0 = __builtin_amdgcn_wmma_f32_16x16x32_bf16(false, Alo, false, Bhi,
                                                       (short)0, acc0, false, false);
      }
      {
        const uint4* rh = (const uint4*)&Shi[(bp + 16) * PROW + bbase];
        const uint4* rl = (const uint4*)&Slo[(bp + 16) * PROW + bbase];
        uint4 x0 = rh[0], x1 = rh[1];
        uint4 y0 = rl[0], y1 = rl[1];
        v8u bh = {x0.x, x0.y, x0.z, x0.w, x1.x, x1.y, x1.z, x1.w};
        v8u bl = {y0.x, y0.y, y0.z, y0.w, y1.x, y1.y, y1.z, y1.w};
        v16bf Bhi = __builtin_bit_cast(v16bf, bh);
        v16bf Blo = __builtin_bit_cast(v16bf, bl);
        acc1 = __builtin_amdgcn_wmma_f32_16x16x32_bf16(false, Ahi, false, Bhi,
                                                       (short)0, acc1, false, false);
        acc1 = __builtin_amdgcn_wmma_f32_16x16x32_bf16(false, Ahi, false, Blo,
                                                       (short)0, acc1, false, false);
        acc1 = __builtin_amdgcn_wmma_f32_16x16x32_bf16(false, Alo, false, Bhi,
                                                       (short)0, acc1, false, false);
      }
    }
    __syncthreads();
  }

  // ---- epilogue: D layout (VGPR r -> M=r / r+8 by lane half) + bias ----
  const int pw0 = p0 + bp;
  const int pw1 = p0 + 16 + bp;
  if (pw0 < NPIX) {
    #pragma unroll
    for (int r = 0; r < 8; ++r) {
      const int cout = m * 16 + bhalf * 8 + r;
      out[((size_t)b * COUT + cout) * NPIX + pw0] = acc0[r] + bias[cout];
    }
  }
  if (pw1 < NPIX) {
    #pragma unroll
    for (int r = 0; r < 8; ++r) {
      const int cout = m * 16 + bhalf * 8 + r;
      out[((size_t)b * COUT + cout) * NPIX + pw1] = acc1[r] + bias[cout];
    }
  }
}

extern "C" void kernel_launch(void* const* d_in, const int* in_sizes, int n_in,
                              void* d_out, int out_size, void* d_ws, size_t ws_size,
                              hipStream_t stream) {
  (void)in_sizes; (void)n_in; (void)out_size; (void)ws_size;
  const float* inp    = (const float*)d_in[0];
  const float* off    = (const float*)d_in[1];
  const float* msk    = (const float*)d_in[2];
  const float* weight = (const float*)d_in[3];
  const float* bias   = (const float*)d_in[4];
  float* out = (float*)d_out;

  // workspace: 512 KB bf16-hi weights + 512 KB bf16-lo weights (1 MB total)
  unsigned short* whi = (unsigned short*)d_ws;
  unsigned short* wlo = whi + (size_t)COUT * CK;

  prep_weights<<<(COUT * CK) / 256, 256, 0, stream>>>(weight, whi, wlo);
  deform_conv_wmma<<<B_N * NTILE, 256, 0, stream>>>(inp, off, msk, bias,
                                                    whi, wlo, out);
}
